// SimilarityTreeLSTM_15857019257422
// MI455X (gfx1250) — matrix operand order repaired and checked by reading, not compiled, not run on hardware
//
#include <hip/hip_runtime.h>
#include <hip/hip_bf16.h>
#include <stdint.h>

// ---------------- problem constants ----------------
#define N_NODES   65536
#define IN_DIM    256
#define MEM_DIM   256
#define IOU_DIM   768          // 3*MEM_DIM
#define BIG_N     1024         // 768 (ioux) + 256 (fx)
#define NT_BIG    64           // BIG_N/16
#define NT_IOU    48           // IOU_DIM/16
#define NT_FH     16           // MEM_DIM/16
#define KT        8            // 256 / 32 k-steps
#define NLEV_LAUNCH 96         // covers tree height w.h.p. (E[h] ~ 30)

typedef __attribute__((ext_vector_type(16))) _Float16 v16h;
typedef __attribute__((ext_vector_type(8)))  float    v8f;

__device__ __forceinline__ float sigf(float x) { return 1.0f / (1.0f + __expf(-x)); }

__device__ __forceinline__ uint32_t pack2h(float a, float b) {
    _Float16 ha = (_Float16)a, hb = (_Float16)b;
    uint16_t ua, ub;
    __builtin_memcpy(&ua, &ha, 2);
    __builtin_memcpy(&ub, &hb, 2);
    return (uint32_t)ua | ((uint32_t)ub << 16);
}

// WMMA 16x16x32 f16 A-layout (wave32):
//   lane = (row&15) + 16*laneHi ; VGPR v(0..3): k = kk*32 + laneHi*8 + 2*(v&3)
//                                 VGPR v(4..7): k = kk*32 + 16 + laneHi*8 + 2*(v&3)
// inverse (k even, row) -> (lane, v):
__device__ __forceinline__ void a_inv(int k, int row, int& lane, int& v) {
    int rem  = k & 31;
    int vhi  = (rem >= 16) ? 1 : 0;
    int rem2 = rem & 15;
    int laneHi = rem2 >> 3;
    v    = vhi * 4 + ((rem2 & 7) >> 1);
    lane = laneHi * 16 + (row & 15);
}

// ---------------- weight packing into WMMA B-layout ----------------
// B-layout (32x16, f16): lane = (n&15) + 16*laneHi; VGPR v: k = kk*32 + (v>=4 ? 16 : 0) + (lane>>4)*8 + 2*(v&3)
__global__ __launch_bounds__(256) void pack_w_kernel(
    const float* __restrict__ Wioux, const float* __restrict__ Wfx,
    const float* __restrict__ Wiouh, const float* __restrict__ Wfh,
    uint32_t* __restrict__ WbigP, uint32_t* __restrict__ WiouhP, uint32_t* __restrict__ WfhP)
{
    int L = blockIdx.x * 256 + threadIdx.x;            // 0 .. 262143
    if (L < 131072) {                                  // Wbig = [Wioux | Wfx], NT=64
        int v = L & 7, lane = (L >> 3) & 31, nt = (L >> 8) & 63, kk = L >> 14;
        int n  = nt * 16 + (lane & 15);
        int k0 = kk * 32 + ((v >= 4) ? 16 : 0) + (lane >> 4) * 8 + 2 * (v & 3);
        float a0 = (n < 768) ? Wioux[(size_t)k0 * 768 + n]       : Wfx[(size_t)k0 * 256 + (n - 768)];
        float a1 = (n < 768) ? Wioux[(size_t)(k0 + 1) * 768 + n] : Wfx[(size_t)(k0 + 1) * 256 + (n - 768)];
        WbigP[L] = pack2h(a0, a1);
    } else if (L < 131072 + 98304) {                   // Wiouh, NT=48
        int L2 = L - 131072;
        int v = L2 & 7, lane = (L2 >> 3) & 31, nt = (L2 >> 8) % 48, kk = L2 / 12288;
        int n  = nt * 16 + (lane & 15);
        int k0 = kk * 32 + ((v >= 4) ? 16 : 0) + (lane >> 4) * 8 + 2 * (v & 3);
        WiouhP[L2] = pack2h(Wiouh[(size_t)k0 * 768 + n], Wiouh[(size_t)(k0 + 1) * 768 + n]);
    } else {                                           // Wfh, NT=16
        int L3 = L - 131072 - 98304;                   // 0..32767
        int v = L3 & 7, lane = (L3 >> 3) & 31, nt = (L3 >> 8) & 15, kk = L3 >> 12;
        int n  = nt * 16 + (lane & 15);
        int k0 = kk * 32 + ((v >= 4) ? 16 : 0) + (lane >> 4) * 8 + 2 * (v & 3);
        WfhP[L3] = pack2h(Wfh[(size_t)k0 * 256 + n], Wfh[(size_t)(k0 + 1) * 256 + n]);
    }
}

// ---------------- embedding gather, packed directly into A-layout ----------------
__global__ __launch_bounds__(256) void pack_x_kernel(
    const int* __restrict__ tok, const float* __restrict__ emb, uint32_t* __restrict__ Xpack)
{
    int tid = blockIdx.x * 256 + threadIdx.x;          // 0 .. 65536*128-1
    int r = tid >> 7, j = tid & 127;
    int k = 2 * j;
    int t = tok[r];
    float v0 = emb[(size_t)t * 256 + k];
    float v1 = emb[(size_t)t * 256 + k + 1];
    int lane, v;
    a_inv(k, r, lane, v);
    int kk = k >> 5, rt = r >> 4;
    Xpack[(((size_t)rt * 8 + kk) * 32 + lane) * 8 + v] = pack2h(v0, v1);
}

// ---------------- big batched GEMM: [65536,256] @ [256,1024] via WMMA ----------------
__global__ __launch_bounds__(256) void gemm_x_kernel(
    const uint32_t* __restrict__ Xpack, const uint32_t* __restrict__ WbigP,
    const float* __restrict__ bioux, const float* __restrict__ bfx,
    float* __restrict__ ioux_all, float* __restrict__ fx_all)
{
    __shared__ uint32_t tA[2048];                      // one 16x256 f16 A tile, packed
    int rt = blockIdx.x;
    const uint32_t* src = Xpack + (size_t)rt * 2048;
    for (int i = threadIdx.x; i < 2048; i += 256) tA[i] = src[i];
    __syncthreads();

    int wave = threadIdx.x >> 5, lane = threadIdx.x & 31;
    v8f acc[8] = {};
    for (int kk = 0; kk < KT; ++kk) {
        v16h a = *(const v16h*)(&tA[(kk * 32 + lane) * 8]);
#pragma unroll
        for (int j = 0; j < 8; ++j) {
            int nt = wave * 8 + j;
            const uint32_t* bp = WbigP + (((size_t)kk * NT_BIG + nt) * 32 + lane) * 8;
            if (j == 0 && kk + 1 < KT)
                __builtin_prefetch(WbigP + (((size_t)(kk + 1) * NT_BIG + nt) * 32 + lane) * 8, 0, 0);
            v16h b = *(const v16h*)bp;
            acc[j] = __builtin_amdgcn_wmma_f32_16x16x32_f16(false, a, false, b, (short)0, acc[j], false, false);
        }
    }
    int colLo = lane & 15, rowHi = (lane >> 4) * 8;
#pragma unroll
    for (int j = 0; j < 8; ++j) {
        int col = (wave * 8 + j) * 16 + colLo;
#pragma unroll
        for (int v = 0; v < 8; ++v) {
            int row = rt * 16 + rowHi + v;
            float val = acc[j][v];
            if (col < 768) ioux_all[(size_t)row * 768 + col] = val + bioux[col];
            else           fx_all[(size_t)row * 256 + (col - 768)] = val + bfx[col - 768];
        }
    }
}

// ---------------- depth by pointer jumping ----------------
__global__ __launch_bounds__(256) void depth_init_kernel(const int* __restrict__ par, int* d, int* a) {
    int i = blockIdx.x * 256 + threadIdx.x;
    d[i] = (i == 0) ? 0 : 1;
    a[i] = (i == 0) ? 0 : par[i];
}
__global__ __launch_bounds__(256) void depth_jump_kernel(const int* __restrict__ din, const int* __restrict__ ain,
                                                         int* dout, int* aout) {
    int i = blockIdx.x * 256 + threadIdx.x;
    int p = ain[i];
    dout[i] = din[i] + din[p];
    aout[i] = ain[p];
}
__global__ __launch_bounds__(256) void maxd_kernel(const int* __restrict__ d, int* maxd) {
    int i = blockIdx.x * 256 + threadIdx.x;
    atomicMax(maxd, d[i]);
}
__global__ __launch_bounds__(256) void hist_kernel(const int* __restrict__ d, int* counts) {
    int i = blockIdx.x * 256 + threadIdx.x;
    atomicAdd(&counts[d[i]], 1);
}
__global__ __launch_bounds__(256) void scan_kernel(const int* __restrict__ counts, int* levoff, int* cursor) {
    __shared__ int part[256];
    __shared__ int pref[257];
    int t = threadIdx.x;
    int s = 0;
    for (int k = 0; k < 256; ++k) s += counts[t * 256 + k];
    part[t] = s;
    __syncthreads();
    if (t == 0) {
        int run = 0;
        for (int i = 0; i < 256; ++i) { pref[i] = run; run += part[i]; }
        pref[256] = run;
    }
    __syncthreads();
    int run = pref[t];
    for (int k = 0; k < 256; ++k) {
        int idx = t * 256 + k;
        levoff[idx] = run; cursor[idx] = run;
        run += counts[idx];
    }
    if (t == 255) levoff[65536] = pref[256];
}
__global__ __launch_bounds__(256) void scatter_kernel(const int* __restrict__ d, int* cursor, int* order) {
    int i = blockIdx.x * 256 + threadIdx.x;
    int pos = atomicAdd(&cursor[d[i]], 1);
    order[pos] = i;
}

// ---------------- one tree level: 16-node tiles, two WMMA GEMMs + gates + parent atomics ----
__global__ __launch_bounds__(256) void level_kernel(
    int j, const int* __restrict__ maxd, const int* __restrict__ levoff,
    const int* __restrict__ order, const int* __restrict__ parent,
    const float* __restrict__ ioux_all, const float* __restrict__ fx_all,
    const float* __restrict__ biouh, const float* __restrict__ bfh,
    float* __restrict__ h_sum, float* __restrict__ fc_sum,
    const uint32_t* __restrict__ WiouhP, const uint32_t* __restrict__ WfhP,
    float* __restrict__ c_root)
{
    int dlev = *maxd - j;
    if (dlev < 0) return;
    int start = levoff[dlev], end = levoff[dlev + 1];
    int cnt = end - start;
    int tiles = (cnt + 15) >> 4;
    if ((int)blockIdx.x >= tiles) return;

    extern __shared__ char smem[];
    uint32_t* ldsA   = (uint32_t*)smem;                        //  8192 B (16x256 f16 A-packed)
    float*    ldsIOU = (float*)(smem + 8192);                  // 49152 B (16x768 f32) / reused as ldsF
    float*    ldsC   = (float*)(smem + 8192 + 49152);          // 16384 B
    float*    ldsH   = (float*)(smem + 8192 + 49152 + 16384);  // 16384 B
    int*      nd     = (int*)(smem + 8192 + 49152 + 32768);    // node[16], par[16]

    int t = threadIdx.x;
    if (t < 16) {
        int g = start + (int)blockIdx.x * 16 + t;
        int node = (g < end) ? order[g] : -1;
        nd[t] = node;
        nd[16 + t] = (node >= 0) ? parent[node] : 0;
    }
    __syncthreads();

    // stage A: gather h_sum rows -> f16 A-layout in LDS (2048 dwords)
    for (int s = 0; s < 8; ++s) {
        int L = t + s * 256;                    // dword index
        int kk = L >> 8, lane = (L >> 3) & 31, v = L & 7;
        int row = lane & 15, laneHi = lane >> 4;
        int k0 = kk * 32 + ((v >= 4) ? 16 : 0) + laneHi * 8 + 2 * (v & 3);
        int node = nd[row];
        float a0 = 0.f, a1 = 0.f;
        if (node >= 0) {
            a0 = h_sum[(size_t)node * 256 + k0];
            a1 = h_sum[(size_t)node * 256 + k0 + 1];
        }
        ldsA[L] = pack2h(a0, a1);
    }
    __syncthreads();

    int wave = t >> 5, lane = t & 31;
    int colLo = lane & 15, rowHi = (lane >> 4) * 8;

    // G1: Hsum(16x256) @ Wiouh(256x768)
    {
        v8f acc[6] = {};
        for (int kk = 0; kk < KT; ++kk) {
            v16h a = *(const v16h*)(&ldsA[(kk * 32 + lane) * 8]);
#pragma unroll
            for (int q = 0; q < 6; ++q) {
                int nt = wave * 6 + q;
                v16h b = *(const v16h*)(WiouhP + (((size_t)kk * NT_IOU + nt) * 32 + lane) * 8);
                acc[q] = __builtin_amdgcn_wmma_f32_16x16x32_f16(false, a, false, b, (short)0, acc[q], false, false);
            }
        }
#pragma unroll
        for (int q = 0; q < 6; ++q) {
            int col = (wave * 6 + q) * 16 + colLo;
#pragma unroll
            for (int v = 0; v < 8; ++v) ldsIOU[(rowHi + v) * 768 + col] = acc[q][v];
        }
    }
    __syncthreads();

    // stage B: gates -> c, h ; repack h as f16 A-layout for G2
    uint16_t* ldsAh = (uint16_t*)ldsA;
    for (int s = 0; s < 16; ++s) {
        int idx = t + s * 256;                  // 0..4095
        int row = idx >> 8, m = idx & 255;
        int node = nd[row];
        float cval = 0.f, hval = 0.f;
        if (node >= 0) {
            size_t nb = (size_t)node * 768;
            float ig = ldsIOU[row * 768 + m]        + ioux_all[nb + m]        + biouh[m];
            float og = ldsIOU[row * 768 + 256 + m]  + ioux_all[nb + 256 + m]  + biouh[256 + m];
            float ug = ldsIOU[row * 768 + 512 + m]  + ioux_all[nb + 512 + m]  + biouh[512 + m];
            cval = sigf(ig) * tanhf(ug) + fc_sum[(size_t)node * 256 + m];
            hval = sigf(og) * tanhf(cval);
        }
        ldsC[row * 256 + m] = cval;
        ldsH[row * 256 + m] = hval;
        int lane2, v2;
        a_inv(m & ~1, row, lane2, v2);
        int kk = m >> 5;
        _Float16 hh = (_Float16)hval;
        uint16_t u; __builtin_memcpy(&u, &hh, 2);
        ldsAh[(((kk * 32 + lane2) * 8) + v2) * 2 + (m & 1)] = u;
    }
    __syncthreads();

    // G2: H(16x256) @ Wfh(256x256) -> ldsF (reuse ldsIOU, 16x256)
    float* ldsF = ldsIOU;
    {
        v8f acc[2] = {};
        for (int kk = 0; kk < KT; ++kk) {
            v16h a = *(const v16h*)(&ldsA[(kk * 32 + lane) * 8]);
#pragma unroll
            for (int q = 0; q < 2; ++q) {
                int nt = wave * 2 + q;
                v16h b = *(const v16h*)(WfhP + (((size_t)kk * NT_FH + nt) * 32 + lane) * 8);
                acc[q] = __builtin_amdgcn_wmma_f32_16x16x32_f16(false, a, false, b, (short)0, acc[q], false, false);
            }
        }
        __syncthreads();   // everyone done reading ldsIOU from stage B? (stage B read it; barrier above covers) – safe order
#pragma unroll
        for (int q = 0; q < 2; ++q) {
            int col = (wave * 2 + q) * 16 + colLo;
#pragma unroll
            for (int v = 0; v < 8; ++v) ldsF[(rowHi + v) * 256 + col] = acc[q][v];
        }
    }
    __syncthreads();

    // stage C: f = sigmoid(G2 + bfh + fx_all[parent]); scatter-add to parent, or emit root c
    for (int s = 0; s < 16; ++s) {
        int idx = t + s * 256;
        int row = idx >> 8, m = idx & 255;
        int node = nd[row];
        if (node < 0) continue;
        float cval = ldsC[row * 256 + m];
        if (node == 0) {
            c_root[m] = cval;
        } else {
            int par = nd[16 + row];
            float f = sigf(ldsF[row * 256 + m] + bfh[m] + fx_all[(size_t)par * 256 + m]);
            atomicAdd(&h_sum[(size_t)par * 256 + m], ldsH[row * 256 + m]);
            atomicAdd(&fc_sum[(size_t)par * 256 + m], f * cval);
        }
    }
}

// ---------------- similarity head ----------------
__global__ __launch_bounds__(256) void head_kernel(
    const float* __restrict__ cl, const float* __restrict__ cr,
    const float* __restrict__ Wh, const float* __restrict__ bh,
    const float* __restrict__ Wp, const float* __restrict__ bp, float* __restrict__ out)
{
    __shared__ float vec[512];
    __shared__ float hid[256];
    int t = threadIdx.x;
    float a = cl[t], b = cr[t];
    vec[t] = a * b;
    vec[256 + t] = fabsf(a - b);
    __syncthreads();
    float s = bh[t];
    for (int k = 0; k < 512; ++k) s += vec[k] * Wh[(size_t)k * 256 + t];
    hid[t] = sigf(s);
    __syncthreads();
    if (t == 0) {
        float l0 = bp[0], l1 = bp[1];
        for (int k = 0; k < 256; ++k) { l0 += hid[k] * Wp[k * 2]; l1 += hid[k] * Wp[k * 2 + 1]; }
        float mx = fmaxf(l0, l1);
        float e0 = __expf(l0 - mx), e1 = __expf(l1 - mx);
        out[0] = e0 / (e0 + e1);
        out[1] = e1 / (e0 + e1);
    }
}

// ---------------- driver ----------------
extern "C" void kernel_launch(void* const* d_in, const int* in_sizes, int n_in,
                              void* d_out, int out_size, void* d_ws, size_t ws_size,
                              hipStream_t stream) {
    const int*   l_tok  = (const int*)d_in[0];
    const int*   l_par  = (const int*)d_in[1];
    const int*   r_tok  = (const int*)d_in[2];
    const int*   r_par  = (const int*)d_in[3];
    const float* emb    = (const float*)d_in[4];
    const float* Wioux  = (const float*)d_in[5];
    const float* bioux  = (const float*)d_in[6];
    const float* Wiouh  = (const float*)d_in[7];
    const float* biouh  = (const float*)d_in[8];
    const float* Wfx    = (const float*)d_in[9];
    const float* bfx    = (const float*)d_in[10];
    const float* Wfh    = (const float*)d_in[11];
    const float* bfh    = (const float*)d_in[12];
    const float* Wh     = (const float*)d_in[13];
    const float* bh     = (const float*)d_in[14];
    const float* Wp     = (const float*)d_in[15];
    const float* bp     = (const float*)d_in[16];
    float* out = (float*)d_out;

    char* p = (char*)d_ws;
    auto take = [&](size_t bytes) -> void* {
        void* r = (void*)p;
        p += (bytes + 255) & ~(size_t)255;
        return r;
    };
    uint32_t* Xpack   = (uint32_t*)take((size_t)N_NODES * 256 * 2);       // 32 MB (f16 packed dwords)
    float*    ioux_a  = (float*)take((size_t)N_NODES * 768 * 4);          // 192 MB
    float*    fx_a    = (float*)take((size_t)N_NODES * 256 * 4);          // 64 MB
    float*    h_sum   = (float*)take((size_t)N_NODES * 256 * 4);          // 64 MB
    float*    fc_sum  = (float*)take((size_t)N_NODES * 256 * 4);          // 64 MB
    uint32_t* WbigP   = (uint32_t*)take(131072 * 4);
    uint32_t* WiouhP  = (uint32_t*)take(98304 * 4);
    uint32_t* WfhP    = (uint32_t*)take(32768 * 4);
    int* d0 = (int*)take(N_NODES * 4);
    int* a0 = (int*)take(N_NODES * 4);
    int* d1 = (int*)take(N_NODES * 4);
    int* a1 = (int*)take(N_NODES * 4);
    int* counts = (int*)take(65536 * 4);
    int* levoff = (int*)take(65537 * 4);
    int* cursor = (int*)take(65536 * 4);
    int* order  = (int*)take(N_NODES * 4);
    int* maxd   = (int*)take(256);
    float* crootL = (float*)take(256 * 4);
    float* crootR = (float*)take(256 * 4);

    pack_w_kernel<<<1024, 256, 0, stream>>>(Wioux, Wfx, Wiouh, Wfh, WbigP, WiouhP, WfhP);

    size_t levLds = 8192 + 49152 + 16384 + 16384 + 128;

    for (int tree = 0; tree < 2; ++tree) {
        const int* tok = (tree == 0) ? l_tok : r_tok;
        const int* par = (tree == 0) ? l_par : r_par;
        float* croot = (tree == 0) ? crootL : crootR;

        pack_x_kernel<<<32768, 256, 0, stream>>>(tok, emb, Xpack);
        gemm_x_kernel<<<4096, 256, 0, stream>>>(Xpack, WbigP, bioux, bfx, ioux_a, fx_a);

        hipMemsetAsync(h_sum,  0, (size_t)N_NODES * 256 * 4, stream);
        hipMemsetAsync(fc_sum, 0, (size_t)N_NODES * 256 * 4, stream);
        hipMemsetAsync(counts, 0, 65536 * 4, stream);
        hipMemsetAsync(maxd,   0, 4, stream);

        depth_init_kernel<<<256, 256, 0, stream>>>(par, d0, a0);
        int *din = d0, *ain = a0, *dout = d1, *aout = a1;
        for (int it = 0; it < 17; ++it) {
            depth_jump_kernel<<<256, 256, 0, stream>>>(din, ain, dout, aout);
            int* tmp = din; din = dout; dout = tmp;
            tmp = ain; ain = aout; aout = tmp;
        }
        maxd_kernel<<<256, 256, 0, stream>>>(din, maxd);
        hist_kernel<<<256, 256, 0, stream>>>(din, counts);
        scan_kernel<<<1, 256, 0, stream>>>(counts, levoff, cursor);
        scatter_kernel<<<256, 256, 0, stream>>>(din, cursor, order);

        for (int j = 0; j < NLEV_LAUNCH; ++j) {
            level_kernel<<<4096, 256, levLds, stream>>>(
                j, maxd, levoff, order, par, ioux_a, fx_a, biouh, bfh,
                h_sum, fc_sum, WiouhP, WfhP, croot);
        }
    }

    head_kernel<<<1, 256, 0, stream>>>(crootL, crootR, Wh, bh, Wp, bp, out);
}
